// DQNAgent_modify_7241314861638
// MI455X (gfx1250) — matrix operand order repaired and checked by reading, not compiled, not run on hardware
//
#include <hip/hip_runtime.h>
#include <hip/hip_bf16.h>

typedef __attribute__((ext_vector_type(16))) __bf16 v16bf;
typedef __attribute__((ext_vector_type(8)))  __bf16 v8bf;
typedef __attribute__((ext_vector_type(8)))  float  v8f;

#define TM   64    // rows per workgroup tile
#define SPAD 520   // LDS activation row stride (bf16 elems): 1040B, 16B-aligned, conflict-free

// ---------------------------------------------------------------------------
// Pack fp32 weight [K][N] row-major into WMMA B-fragment layout (bf16):
// tile t = (k0/32)*(N/16) + n0/16 ; within tile: lane l (0..31) holds 16 bf16
// covering K = k0 + 16*(l/16) + e (e=0..15) at column n0 + (l&15).
// ---------------------------------------------------------------------------
__global__ __launch_bounds__(256) void pack_weight_kernel(
    const float* __restrict__ W, __bf16* __restrict__ out, int Kdim, int Ndim)
{
    int gid = blockIdx.x * 256 + threadIdx.x;
    int total = Kdim * Ndim;
    if (gid >= total) return;
    int t      = gid >> 9;         // /512 elements per 32x16 tile
    int within = gid & 511;
    int l = within >> 4;
    int e = within & 15;
    int tilesN = Ndim >> 4;
    int k0 = (t / tilesN) << 5;
    int n0 = (t % tilesN) << 4;
    int K = k0 + ((l >> 4) << 4) + e;
    int n = n0 + (l & 15);
    out[gid] = (__bf16)W[K * Ndim + n];
}

// ---------------------------------------------------------------------------
// Prep kernel: per-sample fp32 pass.
//   base0[b][j] = (obs[b]/norm) @ W0a   (no bias)
//   qbef[b]    = full fp32 MLP on un-normalized obs[b]
// One workgroup (256 threads) per sample; layers in LDS ping-pong.
// ---------------------------------------------------------------------------
__device__ __forceinline__ void dense_f32(const float* in, float* out,
    const float* __restrict__ W, const float* __restrict__ bias, int K, int N)
{
    int tid = threadIdx.x;
    for (int j = tid; j < N; j += 256) {
        float acc = bias[j];
        for (int k = 0; k < K; ++k) acc = fmaf(in[k], W[k * N + j], acc);
        out[j] = fmaxf(acc, 0.f);
    }
    __syncthreads();
}

__global__ __launch_bounds__(256) void prep_kernel(
    const float* __restrict__ obs,
    const float* __restrict__ W0a, const float* __restrict__ b0a,
    const float* __restrict__ W0b, const float* __restrict__ b0b,
    const float* __restrict__ W1a, const float* __restrict__ b1a,
    const float* __restrict__ W1b, const float* __restrict__ b1b,
    const float* __restrict__ W2a, const float* __restrict__ b2a,
    const float* __restrict__ W2b, const float* __restrict__ b2b,
    const float* __restrict__ Wv,  const float* __restrict__ bv,
    float* __restrict__ base0, float* __restrict__ qbef)
{
    __shared__ float xs[512];   // raw obs (for q_before)
    __shared__ float xn[512];   // normalized obs (for base0)
    __shared__ float ha[512];
    __shared__ float hb[512];
    __shared__ float red[256];

    int b = blockIdx.x, tid = threadIdx.x;
    for (int k = tid; k < 512; k += 256) {
        float v = obs[b * 512 + k];
        xs[k] = v;
        xn[k] = v * ((k & 1) ? (1.f / 160.f) : (1.f / 42.f));
    }
    __syncthreads();

    // base0 (layer-0 partial for perturbed rows; bias added later)
    {
        float acc = 0.f;
        for (int k = 0; k < 512; ++k) acc = fmaf(xn[k], W0a[k * 256 + tid], acc);
        base0[b * 256 + tid] = acc;
    }

    // q_before: full fp32 MLP on un-normalized obs
    dense_f32(xs, ha, W0a, b0a, 512, 256);
    dense_f32(ha, hb, W0b, b0b, 256, 512);
    dense_f32(hb, ha, W1a, b1a, 512, 512);
    dense_f32(ha, hb, W1b, b1b, 512, 256);
    dense_f32(hb, ha, W2a, b2a, 256, 128);
    dense_f32(ha, hb, W2b, b2b, 128, 256);

    red[tid] = hb[tid] * Wv[tid];
    __syncthreads();
    for (int s = 128; s > 0; s >>= 1) {
        if (tid < s) red[tid] += red[tid + s];
        __syncthreads();
    }
    if (tid == 0) qbef[b] = red[0] + bv[0];
}

// ---------------------------------------------------------------------------
// WMMA fused-MLP main kernel. 64 rows per WG, activations in LDS (bf16),
// per-layer GEMM via v_wmma_f32_16x16x32_bf16; each wave owns 16-wide output
// strips and reuses one B fragment across 4 row-blocks.
// ---------------------------------------------------------------------------
__device__ __forceinline__ v16bf load_afrag(const __bf16* ain, int row, int k0, int half)
{
    // A fragment in contiguous k order (same as the B pack): lane holds K = k0+16*half..+15 of row
    const __bf16* p = ain + row * SPAD + k0 + 16 * half;
    v8bf lo = *(const v8bf*)(p);
    v8bf hi = *(const v8bf*)(p + 8);
    return __builtin_shufflevector(lo, hi, 0,1,2,3,4,5,6,7,8,9,10,11,12,13,14,15);
}

__device__ __forceinline__ void wmma_layer(const __bf16* __restrict__ ain,
    __bf16* __restrict__ aout, const __bf16* __restrict__ pw,
    const float* __restrict__ bias, int K, int N)
{
    int tid  = threadIdx.x;
    int lane = tid & 31;
    int wave = tid >> 5;
    int m16  = lane & 15;
    int half = lane >> 4;
    int tilesN = N >> 4;

    for (int nb = wave; nb < tilesN; nb += 8) {
        v8f c0 = {}, c1 = {}, c2 = {}, c3 = {};
        const __bf16* bp = pw + nb * 512 + lane * 16;
        int bstride = tilesN * 512;
        for (int k0 = 0; k0 < K; k0 += 32) {
            v16bf bm = *(const v16bf*)bp;      // one 32B contiguous fragment/lane
            bp += bstride;
            __builtin_prefetch(bp, 0, 1);      // global_prefetch_b8 (next k-tile)
            v16bf a0 = load_afrag(ain,      m16, k0, half);
            v16bf a1 = load_afrag(ain, 16 + m16, k0, half);
            v16bf a2 = load_afrag(ain, 32 + m16, k0, half);
            v16bf a3 = load_afrag(ain, 48 + m16, k0, half);
            c0 = __builtin_amdgcn_wmma_f32_16x16x32_bf16(false, a0, false, bm, (short)0, c0, false, false);
            c1 = __builtin_amdgcn_wmma_f32_16x16x32_bf16(false, a1, false, bm, (short)0, c1, false, false);
            c2 = __builtin_amdgcn_wmma_f32_16x16x32_bf16(false, a2, false, bm, (short)0, c2, false, false);
            c3 = __builtin_amdgcn_wmma_f32_16x16x32_bf16(false, a3, false, bm, (short)0, c3, false, false);
        }
        int   ncol = nb * 16 + m16;
        float bn   = bias[ncol];
#pragma unroll
        for (int v = 0; v < 8; ++v) {
            int mrow = v + 8 * half;           // C layout: VGPR v -> M = v + 8*(lane/16)
            aout[(     mrow) * SPAD + ncol] = (__bf16)fmaxf(c0[v] + bn, 0.f);
            aout[(16 + mrow) * SPAD + ncol] = (__bf16)fmaxf(c1[v] + bn, 0.f);
            aout[(32 + mrow) * SPAD + ncol] = (__bf16)fmaxf(c2[v] + bn, 0.f);
            aout[(48 + mrow) * SPAD + ncol] = (__bf16)fmaxf(c3[v] + bn, 0.f);
        }
    }
    __syncthreads();
}

__global__ __launch_bounds__(256) void mlp_main_kernel(
    const float* __restrict__ feat,
    const float* __restrict__ W0a, const float* __restrict__ b0a,
    const __bf16* __restrict__ pW0b, const float* __restrict__ b0b,
    const __bf16* __restrict__ pW1a, const float* __restrict__ b1a,
    const __bf16* __restrict__ pW1b, const float* __restrict__ b1b,
    const __bf16* __restrict__ pW2a, const float* __restrict__ b2a,
    const __bf16* __restrict__ pW2b, const float* __restrict__ b2b,
    const float* __restrict__ Wv, const float* __restrict__ bv,
    const float* __restrict__ base0, const float* __restrict__ qbef,
    float* __restrict__ out)
{
    extern __shared__ char smemraw[];
    __bf16* actA = (__bf16*)smemraw;           // [TM][SPAD]
    __bf16* actB = actA + TM * SPAD;           // [TM][SPAD]
    float*  rowf = (float*)(actB + TM * SPAD); // [TM][4] pre-scaled feat

    int tid = threadIdx.x;
    int r0  = blockIdx.x * TM;                 // 64-row tile lies within one sample b
    int b   = r0 >> 8;

    if (tid < TM) {
        int rem = (r0 + tid) & 255;
        int kk  = rem & 1;
        const float* f = feat + (b * 2 + kk) * 4;
        rowf[tid * 4 + 0] = f[0] * (1.f / 42.f);   // norm parity depends only on d
        rowf[tid * 4 + 1] = f[1] * (1.f / 160.f);
        rowf[tid * 4 + 2] = f[2] * (1.f / 42.f);
        rowf[tid * 4 + 3] = f[3] * (1.f / 160.f);
    }
    __syncthreads();

    // Layer 0 via rank-4 correction: h0 = relu(base0 - corr + b0a)
    {
        float base = base0[b * 256 + tid] + b0a[tid];
        for (int row = 0; row < TM; ++row) {
            int rem = (r0 + row) & 255;
            int i   = rem >> 1;
            const float* w = W0a + i * 4 * 256 + tid;
            float c = rowf[row * 4 + 0] * w[0]
                    + rowf[row * 4 + 1] * w[256]
                    + rowf[row * 4 + 2] * w[512]
                    + rowf[row * 4 + 3] * w[768];
            actA[row * SPAD + tid] = (__bf16)fmaxf(base - c, 0.f);
        }
    }
    __syncthreads();

    wmma_layer(actA, actB, pW0b, b0b, 256, 512);
    wmma_layer(actB, actA, pW1a, b1a, 512, 512);
    wmma_layer(actA, actB, pW1b, b1b, 512, 256);
    wmma_layer(actB, actA, pW2a, b2a, 256, 128);
    wmma_layer(actA, actB, pW2b, b2b, 128, 256);

    // Final: out[r] = h5 . Wv + bv - q_before[b]   (h5 in actB)
    if (tid < TM) {
        float acc = 0.f;
        const __bf16* h = actB + tid * SPAD;
        for (int n = 0; n < 256; ++n) acc = fmaf((float)h[n], Wv[n], acc);
        out[r0 + tid] = acc + bv[0] - qbef[b];
    }
}

// ---------------------------------------------------------------------------
extern "C" void kernel_launch(void* const* d_in, const int* in_sizes, int n_in,
                              void* d_out, int out_size, void* d_ws, size_t ws_size,
                              hipStream_t stream)
{
    const float* obs  = (const float*)d_in[0];
    const float* feat = (const float*)d_in[1];
    const float* W0a  = (const float*)d_in[2];
    const float* b0a  = (const float*)d_in[3];
    const float* W0b  = (const float*)d_in[4];
    const float* b0b  = (const float*)d_in[5];
    const float* W1a  = (const float*)d_in[6];
    const float* b1a  = (const float*)d_in[7];
    const float* W1b  = (const float*)d_in[8];
    const float* b1b  = (const float*)d_in[9];
    const float* W2a  = (const float*)d_in[10];
    const float* b2a  = (const float*)d_in[11];
    const float* W2b  = (const float*)d_in[12];
    const float* b2b  = (const float*)d_in[13];
    const float* Wv   = (const float*)d_in[14];
    const float* bv   = (const float*)d_in[15];
    float* out = (float*)d_out;

    // Workspace layout
    char* ws = (char*)d_ws;
    __bf16* pW0b = (__bf16*)ws; ws += 256 * 512 * 2;
    __bf16* pW1a = (__bf16*)ws; ws += 512 * 512 * 2;
    __bf16* pW1b = (__bf16*)ws; ws += 512 * 256 * 2;
    __bf16* pW2a = (__bf16*)ws; ws += 256 * 128 * 2;
    __bf16* pW2b = (__bf16*)ws; ws += 128 * 256 * 2;
    float* base0 = (float*)ws;  ws += 512 * 256 * 4;
    float* qbef  = (float*)ws;  ws += 512 * 4;

    // 1) Pack weights into WMMA B layout (bf16)
    auto pack = [&](const float* W, __bf16* o, int K, int N) {
        int total = K * N;
        pack_weight_kernel<<<(total + 255) / 256, 256, 0, stream>>>(W, o, K, N);
    };
    pack(W0b, pW0b, 256, 512);
    pack(W1a, pW1a, 512, 512);
    pack(W1b, pW1b, 512, 256);
    pack(W2a, pW2a, 256, 128);
    pack(W2b, pW2b, 128, 256);

    // 2) base0 + q_before (fp32)
    prep_kernel<<<512, 256, 0, stream>>>(obs, W0a, b0a, W0b, b0b, W1a, b1a,
                                         W1b, b1b, W2a, b2a, W2b, b2b,
                                         Wv, bv, base0, qbef);

    // 3) Main WMMA MLP over 131072 perturbed rows, 64 rows per WG
    size_t smem = (size_t)(2 * TM * SPAD) * sizeof(__bf16) + (size_t)TM * 4 * sizeof(float);
    mlp_main_kernel<<<(512 * 256) / TM, 256, smem, stream>>>(
        feat, W0a, b0a, pW0b, b0b, pW1a, b1a, pW1b, b1b,
        pW2a, b2a, pW2b, b2b, Wv, bv, base0, qbef, out);
}
